// Predictor_82420422410869
// MI455X (gfx1250) — matrix operand order repaired and checked by reading, not compiled, not run on hardware
//
#include <hip/hip_runtime.h>

// RNN-T predictor for MI455X (gfx1250, wave32, WMMA + TDM).
//   B=64, U=256, V=1024, D=1024, J=512
// Pipeline (all on `stream`):
//   1) convert emb/W_ih/W_hh/W_out to bf16 in workspace
//   2) init h-state buffers + global-barrier counter
//   3) gi GEMM  (fused embedding gather, +b_ih)   -> bf16 gi [B,U,3D]
//      64x64 wave tiles: 16 b128 loads : 16 WMMAs (balanced vs L0 bandwidth)
//   4) persistent GRU scan (64 WGs); W_hh slice staged to LDS via the
//      Tensor Data Mover (hardware pad -> bank-conflict-free rows),
//      3 WMMAs/wave/step from LDS, fused gates, global spin barrier
//   5) out GEMM (hs @ W_out^T + b_out)            -> f32 d_out [B,U,J]

#define DDIM 1024
#define B3D  3072
#define BB   64
#define UU   256
#define JJ   512
#define VV   1024
#define SCAN_WGS 64
// TDM pad: 2 DWORDs inserted after every 256 DWORDs -> row stride
// 1024 + 2*4 u16 = 1032 u16 (516 dwords, bank stride 4 -> conflict-free),
// second half of each row shifted by +4 u16.
#define LDS_ROW  1032

typedef __attribute__((ext_vector_type(16))) __bf16 v16bf;
typedef __attribute__((ext_vector_type(8)))  float  v8f;
typedef __attribute__((ext_vector_type(8)))  unsigned short v8us;
typedef __attribute__((ext_vector_type(4)))  unsigned int u32x4;
typedef __attribute__((ext_vector_type(8)))  int i32x8;
typedef __attribute__((ext_vector_type(4)))  int i32x4;

#if defined(__AMDGCN__) && __has_builtin(__builtin_amdgcn_tensor_load_to_lds) && \
    __has_builtin(__builtin_amdgcn_s_wait_tensorcnt)
#define USE_TDM 1
#else
#define USE_TDM 0
#endif

union Frag16 {
  v16bf bf;
  v8us  h[2];
};

__device__ __forceinline__ unsigned short f2bf(float x) {
  unsigned u = __float_as_uint(x);
  u += 0x7fffu + ((u >> 16) & 1u);          // round-to-nearest-even
  return (unsigned short)(u >> 16);
}
__device__ __forceinline__ float bf2f(unsigned short s) {
  return __uint_as_float(((unsigned)s) << 16);
}
__device__ __forceinline__ float sigmoid_f(float x) {
  return 1.0f / (1.0f + __expf(-x));
}

// A fragment (16x32 MxK bf16): per lane row = lane%16,
// 16 contiguous K (pointer already includes +16*(lane/16) K offset)
__device__ __forceinline__ v16bf load_a_frag(const unsigned short* p) {
  Frag16 f;
  f.h[0] = *(const v8us*)(p);
  f.h[1] = *(const v8us*)(p + 8);
  return f.bf;
}
// B fragment (32x16 KxN bf16): per lane col = lane%16, 16 contiguous K
// (pointer already includes +16*(lane/16) K offset)
__device__ __forceinline__ v16bf load_b_frag(const unsigned short* p) {
  Frag16 f;
  f.h[0] = *(const v8us*)(p);
  f.h[1] = *(const v8us*)(p + 8);
  return f.bf;
}
__device__ __forceinline__ v8f wmma_bf16(v16bf a, v16bf b, v8f c) {
  return __builtin_amdgcn_wmma_f32_16x16x32_bf16(false, a, false, b,
                                                 (short)0, c, false, false);
}

// ---------------------------------------------------------------- converts
__global__ void cvt_f32_bf16_kernel(const float* __restrict__ src,
                                    unsigned short* __restrict__ dst, int n) {
  int i = blockIdx.x * blockDim.x + threadIdx.x;
  if (i < n) dst[i] = f2bf(src[i]);
}

__global__ void init_state_kernel(const float* __restrict__ h0,
                                  float* __restrict__ h32a,
                                  unsigned short* __restrict__ h0b,
                                  unsigned int* __restrict__ counter) {
  int i = blockIdx.x * blockDim.x + threadIdx.x;
  if (i == 0) *counter = 0u;
  if (i < BB * DDIM) {
    float v = h0[i & (DDIM - 1)];
    h32a[i] = v;
    h0b[i]  = f2bf(v);
  }
}

// ------------------------------------------------------- gi = emb[idx]@Wih^T
// grid (3072/128, 16384/256), 256 thr = 8 waves (4M x 2N), wave tile 64x64
__global__ void __launch_bounds__(256)
gi_gemm_kernel(const int* __restrict__ y,                // [64][256]
               const unsigned short* __restrict__ emb_b, // [1024][1024]
               const unsigned short* __restrict__ Wih_b, // [3072][1024]
               const float* __restrict__ b_ih,           // [3072]
               unsigned short* __restrict__ gi_b)        // [16384][3072]
{
  const int tid  = threadIdx.x;
  const int lane = tid & 31, wv = tid >> 5;
  const int hf   = lane >> 4, lr = lane & 15;
  const int m_wave = blockIdx.y * 256 + (wv >> 1) * 64;
  const int n_wave = blockIdx.x * 128 + (wv & 1) * 64;

  const unsigned short* pa[4];
#pragma unroll
  for (int t = 0; t < 4; ++t) {
    int m = m_wave + t * 16 + lr;
    int b = m >> 8, u = m & 255;
    int tok = y[(b << 8) | ((u - 1) & 255)];     // jnp.roll(y,1,axis=1)
    pa[t] = emb_b + (size_t)tok * DDIM + hf * 16;
  }
  const unsigned short* pb[4];
#pragma unroll
  for (int t = 0; t < 4; ++t)
    pb[t] = Wih_b + (size_t)(n_wave + t * 16 + lr) * DDIM + hf * 16;

  v8f acc[4][4] = {};
  for (int k = 0; k < DDIM; k += 32) {
    v16bf af[4], bfr[4];
#pragma unroll
    for (int t = 0; t < 4; ++t) af[t]  = load_a_frag(pa[t] + k);
#pragma unroll
    for (int t = 0; t < 4; ++t) bfr[t] = load_b_frag(pb[t] + k);
#pragma unroll
    for (int tm = 0; tm < 4; ++tm)
#pragma unroll
      for (int tn = 0; tn < 4; ++tn)
        acc[tm][tn] = wmma_bf16(af[tm], bfr[tn], acc[tm][tn]);
  }
#pragma unroll
  for (int tm = 0; tm < 4; ++tm)
#pragma unroll
    for (int tn = 0; tn < 4; ++tn) {
      int n = n_wave + tn * 16 + lr;
      float bias = b_ih[n];
      int m0 = m_wave + tm * 16 + hf * 8;
#pragma unroll
      for (int g = 0; g < 8; ++g)
        gi_b[(size_t)(m0 + g) * B3D + n] = f2bf(acc[tm][tn][g] + bias);
    }
}

// ----------------------------------------------------- persistent GRU scan
// 64 WGs x 128 thr (4 waves). WG owns d-columns [16*bid, 16*bid+16).
__global__ void __launch_bounds__(128)
scan_kernel(const unsigned short* __restrict__ Whh_b, // [3072][1024]
            const unsigned short* __restrict__ gi_b,  // [16384][3072]
            const float* __restrict__ b_hh,           // [3072]
            const unsigned short* __restrict__ h0b,   // [64][1024]
            float* __restrict__ h32a, float* __restrict__ h32b,
            unsigned short* __restrict__ hs_b,        // [256][64][1024]
            unsigned int* __restrict__ counter)
{
  extern __shared__ unsigned short lds_w[];           // [48][LDS_ROW]
  const int tid  = threadIdx.x;
  const int lane = tid & 31, wv = tid >> 5;
  const int hf   = lane >> 4, lr = lane & 15;
  const int d0   = blockIdx.x * 16;

#if USE_TDM
  // Stage 3x (16 rows x 1024 bf16) of W_hh into LDS with the Tensor Data
  // Mover; hardware inserts 2 DWORDs of pad per 256 DWORDs (row stride
  // 1032 u16, second row-half shifted +4 u16).
  if (wv == 0) {
    unsigned lds_base = (unsigned)(uintptr_t)(&lds_w[0]);
#pragma unroll
    for (int g = 0; g < 3; ++g) {
      unsigned long long ga = (unsigned long long)(uintptr_t)(
          Whh_b + (size_t)(g * DDIM + d0) * DDIM);
      u32x4 g0;
      g0[0] = 1u;                                        // count=1, user D#
      g0[1] = lds_base + (unsigned)(g * 16 * LDS_ROW * 2); // lds_addr
      g0[2] = (unsigned)ga;                              // global_addr lo
      g0[3] = (unsigned)((ga >> 32) & 0x1ffffffu) | (2u << 30); // hi | type=2
      i32x8 g1;
      g1[0] = (int)((1u << 16)      // data_size = 2 bytes
                  | (1u << 20)      // pad_enable
                  | (7u << 22)      // pad_interval: every 256 DWORDs
                  | (1u << 25));    // pad_amount: 2 DWORDs
      g1[1] = (int)(1024u << 16);   // tensor_dim0 = 1024 (bits 79:48)
      g1[2] = (int)(16u << 16);     // tensor_dim1 = 16   (bits 111:80)
      g1[3] = (int)(1024u << 16);   // tile_dim0 = 1024   (bits 127:112)
      g1[4] = 16;                   // tile_dim1 = 16
      g1[5] = 1024;                 // tensor_dim0_stride = 1024
      g1[6] = 0;
      g1[7] = 0;
      i32x4 g2; g2[0] = g2[1] = g2[2] = g2[3] = 0;
      i32x4 g3; g3[0] = g3[1] = g3[2] = g3[3] = 0;
      i32x8 g4; g4[0] = g4[1] = g4[2] = g4[3] = 0;
      g4[4] = g4[5] = g4[6] = g4[7] = 0;
      // 6-arg toolchain form: (g0, g1, g2, g3, g4, cpol)
      __builtin_amdgcn_tensor_load_to_lds(g0, g1, g2, g3, g4, 0);
    }
    __builtin_amdgcn_s_wait_tensorcnt(0);
  }
#else
  // Fallback: manual copy replicating the padded layout.
  for (int c = tid; c < 48 * 128; c += 128) {
    int row = c >> 7, cc = c & 127;
    int g = row >> 4, j = row & 15;
    int e = cc * 8;
    *(v8us*)(lds_w + row * LDS_ROW + e + ((e >= 512) ? 4 : 0)) =
        *(const v8us*)(Whh_b + ((size_t)(g * DDIM + d0 + j)) * DDIM + e);
  }
#endif
  __syncthreads();

  const int d = d0 + lr;
  const float bhr = b_hh[d], bhz = b_hh[DDIM + d], bhn = b_hh[2 * DDIM + d];
  const unsigned short* pb0 = lds_w + (0 * 16 + lr) * LDS_ROW + hf * 16;
  const unsigned short* pb1 = lds_w + (1 * 16 + lr) * LDS_ROW + hf * 16;
  const unsigned short* pb2 = lds_w + (2 * 16 + lr) * LDS_ROW + hf * 16;
  float* hbuf[2] = { h32a, h32b };
  const int mbase = wv * 16 + hf * 8;

  for (int u = 0; u < UU; ++u) {
    const unsigned short* Aprev =
        (u == 0) ? h0b : (hs_b + (size_t)(u - 1) * BB * DDIM);
    const unsigned short* pa = Aprev + (size_t)(wv * 16 + lr) * DDIM + hf * 16;

    // prefetch gate inputs so their latency hides under the WMMA loop
    const float* hprev = hbuf[u & 1];
    unsigned short giR[8], giZ[8], giN[8];
    float hpv[8];
#pragma unroll
    for (int g = 0; g < 8; ++g) {
      int m = mbase + g;
      size_t gio = ((size_t)(m * UU + u)) * B3D + d;
      giR[g] = gi_b[gio];
      giZ[g] = gi_b[gio + DDIM];
      giN[g] = gi_b[gio + 2 * DDIM];
      hpv[g] = hprev[m * DDIM + d];
    }

    v8f accR = {}, accZ = {}, accN = {};
#pragma unroll 4
    for (int k = 0; k < 512; k += 32) {     // first row-half (no pad shift)
      v16bf a  = load_a_frag(pa + k);
      accR = wmma_bf16(a, load_b_frag(pb0 + k), accR);
      accZ = wmma_bf16(a, load_b_frag(pb1 + k), accZ);
      accN = wmma_bf16(a, load_b_frag(pb2 + k), accN);
    }
#pragma unroll 4
    for (int k = 512; k < 1024; k += 32) {  // second row-half (+4 u16 pad)
      v16bf a  = load_a_frag(pa + k);
      accR = wmma_bf16(a, load_b_frag(pb0 + k + 4), accR);
      accZ = wmma_bf16(a, load_b_frag(pb1 + k + 4), accZ);
      accN = wmma_bf16(a, load_b_frag(pb2 + k + 4), accN);
    }

    float* hcur = hbuf[(u + 1) & 1];
#pragma unroll
    for (int g = 0; g < 8; ++g) {
      int m = mbase + g;
      float r  = sigmoid_f(bf2f(giR[g]) + accR[g] + bhr);
      float z  = sigmoid_f(bf2f(giZ[g]) + accZ[g] + bhz);
      float nn = tanhf(bf2f(giN[g]) + r * (accN[g] + bhn));
      float hn = (1.0f - z) * nn + z * hpv[g];
      hcur[m * DDIM + d] = hn;
      hs_b[(size_t)(u * BB + m) * DDIM + d] = f2bf(hn);
    }

    // device-wide barrier between steps
    __syncthreads();
    __threadfence();
    if (tid == 0) {
      __atomic_fetch_add(counter, 1u, __ATOMIC_ACQ_REL);
      unsigned target = (unsigned)(u + 1) * SCAN_WGS;
      while (__atomic_load_n(counter, __ATOMIC_ACQUIRE) < target)
        __builtin_amdgcn_s_sleep(1);
    }
    __syncthreads();
    __threadfence();   // invalidate L0 so all lanes see other WGs' h writes
  }
}

// ------------------------------------------------------- out = hs@Wout^T+b
// grid (512/128, 16384/256); A rows are m' = u*64 + b
__global__ void __launch_bounds__(256)
out_gemm_kernel(const unsigned short* __restrict__ hs_b,  // [16384][1024]
                const unsigned short* __restrict__ Wout_b,// [512][1024]
                const float* __restrict__ b_out,          // [512]
                float* __restrict__ out)                  // [64][256][512]
{
  const int tid  = threadIdx.x;
  const int lane = tid & 31, wv = tid >> 5;
  const int hf   = lane >> 4, lr = lane & 15;
  const int m_wave = blockIdx.y * 256 + (wv >> 1) * 64;
  const int n_wave = blockIdx.x * 128 + (wv & 1) * 64;

  const unsigned short* pa[4];
#pragma unroll
  for (int t = 0; t < 4; ++t)
    pa[t] = hs_b + (size_t)(m_wave + t * 16 + lr) * DDIM + hf * 16;
  const unsigned short* pb[4];
#pragma unroll
  for (int t = 0; t < 4; ++t)
    pb[t] = Wout_b + (size_t)(n_wave + t * 16 + lr) * DDIM + hf * 16;

  v8f acc[4][4] = {};
  for (int k = 0; k < DDIM; k += 32) {
    v16bf af[4], bfr[4];
#pragma unroll
    for (int t = 0; t < 4; ++t) af[t]  = load_a_frag(pa[t] + k);
#pragma unroll
    for (int t = 0; t < 4; ++t) bfr[t] = load_b_frag(pb[t] + k);
#pragma unroll
    for (int tm = 0; tm < 4; ++tm)
#pragma unroll
      for (int tn = 0; tn < 4; ++tn)
        acc[tm][tn] = wmma_bf16(af[tm], bfr[tn], acc[tm][tn]);
  }
#pragma unroll
  for (int tm = 0; tm < 4; ++tm)
#pragma unroll
    for (int tn = 0; tn < 4; ++tn) {
      int n = n_wave + tn * 16 + lr;
      float bias = b_out[n];
      int m0 = m_wave + tm * 16 + hf * 8;
#pragma unroll
      for (int g = 0; g < 8; ++g) {
        int m = m0 + g;                    // m = u*64 + b
        int u = m >> 6, b = m & 63;
        out[((size_t)(b << 8 | u)) * JJ + n] = acc[tm][tn][g] + bias;
      }
    }
}

// ---------------------------------------------------------------- launcher
extern "C" void kernel_launch(void* const* d_in, const int* in_sizes, int n_in,
                              void* d_out, int out_size, void* d_ws, size_t ws_size,
                              hipStream_t stream) {
  (void)in_sizes; (void)n_in; (void)out_size; (void)ws_size;
  const int*   y     = (const int*)  d_in[0];
  const float* emb   = (const float*)d_in[1];
  const float* W_ih  = (const float*)d_in[2];
  const float* b_ih  = (const float*)d_in[3];
  const float* W_hh  = (const float*)d_in[4];
  const float* b_hh  = (const float*)d_in[5];
  const float* W_out = (const float*)d_in[6];
  const float* b_out = (const float*)d_in[7];
  const float* h0    = (const float*)d_in[8];
  float* out = (float*)d_out;

  uintptr_t base = (uintptr_t)d_ws;
  size_t off = 0;
  auto carve = [&](size_t bytes) -> uintptr_t {
    uintptr_t p = base + off;
    off = (off + bytes + 255) & ~(size_t)255;
    return p;
  };
  unsigned short* emb_b  = (unsigned short*)carve((size_t)VV * DDIM * 2);
  unsigned short* Wih_b  = (unsigned short*)carve((size_t)B3D * DDIM * 2);
  unsigned short* Whh_b  = (unsigned short*)carve((size_t)B3D * DDIM * 2);
  unsigned short* Wout_b = (unsigned short*)carve((size_t)JJ * DDIM * 2);
  unsigned short* gi_b   = (unsigned short*)carve((size_t)BB * UU * B3D * 2);
  unsigned short* hs_b   = (unsigned short*)carve((size_t)UU * BB * DDIM * 2);
  float*          h32a   = (float*)carve((size_t)BB * DDIM * 4);
  float*          h32b   = (float*)carve((size_t)BB * DDIM * 4);
  unsigned short* h0b    = (unsigned short*)carve((size_t)BB * DDIM * 2);
  unsigned int*   ctr    = (unsigned int*)carve(256);

  auto cvt = [&](const float* s, unsigned short* dst, int n) {
    cvt_f32_bf16_kernel<<<(n + 255) / 256, 256, 0, stream>>>(s, dst, n);
  };
  cvt(emb,   emb_b,  VV * DDIM);
  cvt(W_ih,  Wih_b,  B3D * DDIM);
  cvt(W_hh,  Whh_b,  B3D * DDIM);
  cvt(W_out, Wout_b, JJ * DDIM);

  init_state_kernel<<<(BB * DDIM + 255) / 256, 256, 0, stream>>>(h0, h32a, h0b, ctr);

  gi_gemm_kernel<<<dim3(B3D / 128, (BB * UU) / 256), 256, 0, stream>>>(
      y, emb_b, Wih_b, b_ih, gi_b);

  scan_kernel<<<SCAN_WGS, 128, 48 * LDS_ROW * 2, stream>>>(
      Whh_b, gi_b, b_hh, h0b, h32a, h32b, hs_b, ctr);

  out_gemm_kernel<<<dim3(JJ / 128, (BB * UU) / 256), 256, 0, stream>>>(
      hs_b, Wout_b, b_out, out);
}
